// OptimizedLinear_88252987998616
// MI455X (gfx1250) — matrix operand order repaired and checked by reading, not compiled, not run on hardware
//
#include <hip/hip_runtime.h>

// ---------------------------------------------------------------------------
// Fused per-sample-LR linear layer for MI455X (gfx1250, wave32, WMMA).
//   out = (X @ W^T + bias) - lr[:,None] * (X @ G^T + bias_grad)
// B=4096, K=2048, OUT=2048, fp32 in/out.
//
// base GEMM: fp32 emulated via 3x BF16 WMMA (hi/lo split).
// pert GEMM: single BF16 WMMA (its contribution is scaled by lr <= 1e-3, so
//            bf16 rounding lands ~1e-6 absolute on an O(1) output).
//
// Block = 256 thr (8 waves), tile 128(M)x64(N), KT=32, double-buffered LDS
// with XOR-swizzled stride-32 rows (conflict-free b128 fragment reads).
// ---------------------------------------------------------------------------

typedef __attribute__((ext_vector_type(16))) __bf16 v16bf;
typedef __attribute__((ext_vector_type(8)))  float  v8f;

#define B_DIM 4096
#define K_DIM 2048
#define N_DIM 2048
#define TM    128         // block tile rows (M)
#define TN    64          // block tile cols (N)
#define KT    32          // K-slab per iteration (one bf16 WMMA K-step)

// POD 128-bit / float4 helpers (HIP's uint4/float4 have ctors -> not union-safe)
struct alignas(16) U128 { unsigned u0, u1, u2, u3; };
struct alignas(16) F4   { float x, y, z, w; };

// Element offset of 16B chunk `c` in a swizzled stride-32 bf16 row.
// chunk' = c ^ ((row>>2)&3) makes lanes 0..15 (rows r..r+15) hit disjoint
// bank groups for ds_load_b128 / ds_store_b128.
__device__ __forceinline__ int swz_off(int row, int c) {
  return row * KT + ((c ^ ((row >> 2) & 3)) << 3);
}

__device__ __forceinline__ unsigned pack2(__bf16 a, __bf16 b) {
  union { __bf16 h[2]; unsigned u; } cv;
  cv.h[0] = a; cv.h[1] = b;
  return cv.u;
}

// 8 f32 (from regs) -> bf16 hi/lo planes in LDS (two ds_store_b128).
__device__ __forceinline__ void cvt8_hilo(F4 a, F4 b,
                                          unsigned short* dhi,
                                          unsigned short* dlo,
                                          int row, int chunk) {
  float xs[8] = {a.x, a.y, a.z, a.w, b.x, b.y, b.z, b.w};
  unsigned hU[4], lU[4];
#pragma unroll
  for (int j = 0; j < 4; ++j) {
    float p = xs[2 * j], q = xs[2 * j + 1];
    __bf16 ph = (__bf16)p, qh = (__bf16)q;
    __bf16 pl = (__bf16)(p - (float)ph);
    __bf16 ql = (__bf16)(q - (float)qh);
    hU[j] = pack2(ph, qh);
    lU[j] = pack2(pl, ql);
  }
  U128 hq = {hU[0], hU[1], hU[2], hU[3]};
  U128 lq = {lU[0], lU[1], lU[2], lU[3]};
  *(U128*)&dhi[swz_off(row, chunk)] = hq;
  *(U128*)&dlo[swz_off(row, chunk)] = lq;
}

// 8 f32 -> bf16 hi plane only (one ds_store_b128).
__device__ __forceinline__ void cvt8_hi(F4 a, F4 b,
                                        unsigned short* dhi,
                                        int row, int chunk) {
  float xs[8] = {a.x, a.y, a.z, a.w, b.x, b.y, b.z, b.w};
  unsigned hU[4];
#pragma unroll
  for (int j = 0; j < 4; ++j)
    hU[j] = pack2((__bf16)xs[2 * j], (__bf16)xs[2 * j + 1]);
  U128 hq = {hU[0], hU[1], hU[2], hU[3]};
  *(U128*)&dhi[swz_off(row, chunk)] = hq;
}

// A-matrix 16x32 bf16 fragment (matches B): lanes 0-15 -> K0..15 in v0-7;
// lanes 16-31 -> K16..31.  kbA = (lane>=16)*8.
__device__ __forceinline__ v16bf ldfragA(const unsigned short* p, int row, int kbA) {
  const int c0 = kbA >> 2;              // 0 or 2
  union { v16bf v; U128 q[2]; } u;
  u.q[0] = *(const U128*)(p + swz_off(row, c0));
  u.q[1] = *(const U128*)(p + swz_off(row, c0 + 1));
  return u.v;
}

// B-matrix 32x16 bf16 fragment: lanes 0-15 hold K=0..15 in v0-7,
// lanes 16-31 hold K=16..31.  kbB = (lane>=16)*16.
__device__ __forceinline__ v16bf ldfragB(const unsigned short* p, int row, int kbB) {
  const int c0 = kbB >> 3;              // 0 or 2
  union { v16bf v; U128 q[2]; } u;
  u.q[0] = *(const U128*)(p + swz_off(row, c0));
  u.q[1] = *(const U128*)(p + swz_off(row, c0 + 1));
  return u.v;
}

__device__ __forceinline__ v8f wmma_bf16(v16bf a, v16bf b, v8f c) {
  // (neg_a, A, neg_b, B, c_mod, C, reuse_a, reuse_b)
  return __builtin_amdgcn_wmma_f32_16x16x32_bf16(false, a, false, b,
                                                 (short)0, c, false, false);
}

__global__ __launch_bounds__(256)
void fused_lr_linear_bf16_pipe_kernel(const float* __restrict__ X,
                                      const float* __restrict__ W,
                                      const float* __restrict__ bias,
                                      const float* __restrict__ G,
                                      const float* __restrict__ bgrad,
                                      const float* __restrict__ lr,
                                      float* __restrict__ out) {
  // Double-buffered bf16 planes, swizzled stride-32 rows.  ~57 KB total.
  __shared__ unsigned short sXhi[2][TM * KT];
  __shared__ unsigned short sXlo[2][TM * KT];
  __shared__ unsigned short sWhi[2][TN * KT];
  __shared__ unsigned short sWlo[2][TN * KT];
  __shared__ unsigned short sGhi[2][TN * KT];
  __shared__ float sLr[TM];
  __shared__ float sBias[TN];
  __shared__ float sBg[TN];

  const int tid   = threadIdx.x;
  const int lane  = tid & 31;            // wave32
  const int wave  = tid >> 5;            // 0..7
  const int laneR = lane & 15;
  const int kbA   = (lane >> 4) << 3;    // 0 or 8
  const int kbB   = (lane >> 4) << 4;    // 0 or 16

  const int m0 = blockIdx.y * TM;
  const int n0 = blockIdx.x * TN;

  // Per-wave 32x32 sub-tile (2x2 of 16x16).
  const int mBase = (wave >> 1) * 32;    // 0,32,64,96
  const int nBase = (wave & 1) * 32;     // 0 or 32

  // ---- staging assignments: 4 threads/row, 8 f32 per chunk -----------------
  const int rowX0 = tid >> 2;            // 0..63
  const int rowX1 = 64 + (tid >> 2);     // 64..127
  const int rowW  = tid >> 2;            // 0..63 (also G)
  const int chk   = tid & 3;             // chunk 0..3 (8 f32 each)

  const float* pX0 = X + (size_t)(m0 + rowX0) * K_DIM + chk * 8;
  const float* pX1 = X + (size_t)(m0 + rowX1) * K_DIM + chk * 8;
  const float* pW  = W + (size_t)(n0 + rowW)  * K_DIM + chk * 8;
  const float* pG  = G + (size_t)(n0 + rowW)  * K_DIM + chk * 8;

  // Slab staging registers (32 f32).
  F4 x0a, x0b, x1a, x1b, wa, wb, ga, gb;

#define LOAD_SLAB(K0)                                                        \
  do {                                                                       \
    x0a = *(const F4*)(pX0 + (K0));  x0b = *(const F4*)(pX0 + (K0) + 4);     \
    x1a = *(const F4*)(pX1 + (K0));  x1b = *(const F4*)(pX1 + (K0) + 4);     \
    wa  = *(const F4*)(pW  + (K0));  wb  = *(const F4*)(pW  + (K0) + 4);     \
    ga  = *(const F4*)(pG  + (K0));  gb  = *(const F4*)(pG  + (K0) + 4);     \
  } while (0)

#define CVT_STORE_SLAB(BUF)                                                  \
  do {                                                                       \
    cvt8_hilo(x0a, x0b, sXhi[BUF], sXlo[BUF], rowX0, chk);                   \
    cvt8_hilo(x1a, x1b, sXhi[BUF], sXlo[BUF], rowX1, chk);                   \
    cvt8_hilo(wa,  wb,  sWhi[BUF], sWlo[BUF], rowW,  chk);                   \
    cvt8_hi  (ga,  gb,  sGhi[BUF],            rowW,  chk);                   \
  } while (0)

  // Prologue: stage slab 0, plus lr/bias tables.
  LOAD_SLAB(0);
  if (tid < 128)      sLr[tid]         = lr[m0 + tid];
  else if (tid < 192) sBias[tid - 128] = bias[n0 + tid - 128];
  else                sBg[tid - 192]   = bgrad[n0 + tid - 192];
  CVT_STORE_SLAB(0);
  __syncthreads();

  v8f accB[2][2] = {};                   // base accumulators [mt][nt]
  v8f accP[2][2] = {};                   // pert accumulators [mt][nt]

  int cur = 0;
  for (int k0 = 0; k0 < K_DIM; k0 += KT) {
    const bool hasNext = (k0 + KT) < K_DIM;
    if (hasNext) {
      __builtin_prefetch(pX0 + k0 + 2 * KT, 0, 3);   // global_prefetch_b8
      __builtin_prefetch(pW  + k0 + 2 * KT, 0, 3);
      __builtin_prefetch(pG  + k0 + 2 * KT, 0, 3);
      LOAD_SLAB(k0 + KT);                // issue globals before the WMMAs
    }

    // ---- compute from buffer `cur` ----------------------------------------
    {
      const unsigned short* xh = sXhi[cur];
      const unsigned short* xl = sXlo[cur];
      const unsigned short* wh = sWhi[cur];
      const unsigned short* wl = sWlo[cur];
      const unsigned short* gh = sGhi[cur];

      v16bf aHi[2], aLo[2];
#pragma unroll
      for (int mt = 0; mt < 2; ++mt) {
        const int r = mBase + mt * 16 + laneR;
        aHi[mt] = ldfragA(xh, r, kbA);
        aLo[mt] = ldfragA(xl, r, kbA);
      }
#pragma unroll
      for (int nt = 0; nt < 2; ++nt) {
        const int r = nBase + nt * 16 + laneR;
        v16bf bH = ldfragB(wh, r, kbB);
        v16bf bL = ldfragB(wl, r, kbB);
        v16bf gH = ldfragB(gh, r, kbB);
#pragma unroll
        for (int mt = 0; mt < 2; ++mt) {
          accB[mt][nt] = wmma_bf16(aHi[mt], bH, accB[mt][nt]);
          accB[mt][nt] = wmma_bf16(aHi[mt], bL, accB[mt][nt]);
          accB[mt][nt] = wmma_bf16(aLo[mt], bH, accB[mt][nt]);
          accP[mt][nt] = wmma_bf16(aHi[mt], gH, accP[mt][nt]);
        }
      }
    }

    if (hasNext) CVT_STORE_SLAB(cur ^ 1);
    __syncthreads();                     // one barrier per iteration
    cur ^= 1;
  }

  // ---- epilogue. C/D f32 16x16 layout: lanes 0-15: N=lane, VGPR r -> M=r;
  // lanes 16-31: N=lane-16, VGPR r -> M=8+r.
  const int rowHalf = (lane >> 4) << 3;
#pragma unroll
  for (int mt = 0; mt < 2; ++mt) {
#pragma unroll
    for (int nt = 0; nt < 2; ++nt) {
      const int colL = nBase + nt * 16 + laneR;
      const int colG = n0 + colL;
      const float bc = sBias[colL];
      const float bg = sBg[colL];
#pragma unroll
      for (int r = 0; r < 8; ++r) {
        const int rowL = mBase + mt * 16 + rowHalf + r;
        const float l  = sLr[rowL];
        const float v  = accB[mt][nt][r] + bc - l * (accP[mt][nt][r] + bg);
        out[(size_t)(m0 + rowL) * N_DIM + colG] = v;
      }
    }
  }
#undef LOAD_SLAB
#undef CVT_STORE_SLAB
}

extern "C" void kernel_launch(void* const* d_in, const int* in_sizes, int n_in,
                              void* d_out, int out_size, void* d_ws, size_t ws_size,
                              hipStream_t stream) {
  (void)in_sizes; (void)n_in; (void)out_size; (void)d_ws; (void)ws_size;
  const float* X     = (const float*)d_in[0];   // [4096, 2048]
  const float* W     = (const float*)d_in[1];   // [2048, 2048]
  const float* bias  = (const float*)d_in[2];   // [2048]
  const float* G     = (const float*)d_in[3];   // [2048, 2048]
  const float* bgrad = (const float*)d_in[4];   // [2048]
  const float* lr    = (const float*)d_in[5];   // [4096]
  float* out = (float*)d_out;                   // [4096, 2048]

  dim3 grid(N_DIM / TN, B_DIM / TM);            // 32 x 32
  fused_lr_linear_bf16_pipe_kernel<<<grid, 256, 0, stream>>>(X, W, bias, G,
                                                             bgrad, lr, out);
}
